// VectorQuantizer_61392262529630
// MI455X (gfx1250) — matrix operand order repaired and checked by reading, not compiled, not run on hardware
//
#include <hip/hip_runtime.h>

// ---- CDNA5 (gfx1250, wave32) VQ-VAE vector quantizer -----------------------
// d = ||e_k||^2 - 2 z.e_k  (||z||^2 dropped: constant per row for argmin)
// GEMM via v_wmma_f32_16x16x32_bf16 with fp32 -> bf16 hi/lo split:
//   dot ~= hi.hi + hi.lo + lo.hi   (effective ~16-bit mantissa accuracy)

typedef __attribute__((ext_vector_type(16))) __bf16 v16bf;
typedef __attribute__((ext_vector_type(8)))  __bf16 v8bf;
typedef __attribute__((ext_vector_type(8)))  float  v8f;

#define N_CODES   16384
#define EMB_DIM   256
#define N_VEC     8192              // 8 * 32 * 32
#define ZQ_ELEMS  2097152           // 8 * 256 * 32 * 32
#define NBLK      512               // N_VEC / 16 rows per block
#define CODES_PER_WAVE (N_CODES / 8)

// ---------------------------------------------------------------------------
// Prep: emb fp32 -> bf16 hi/lo + ||e||^2 per code. One block per code.
__global__ __launch_bounds__(256) void vq_prep_emb(
    const float* __restrict__ emb,
    __bf16* __restrict__ ehi, __bf16* __restrict__ elo,
    float* __restrict__ enorm) {
  __shared__ float red[256];
  const int c = blockIdx.x;
  const int d = threadIdx.x;
  const float e = emb[c * EMB_DIM + d];
  const __bf16 h = (__bf16)e;
  ehi[c * EMB_DIM + d] = h;
  elo[c * EMB_DIM + d] = (__bf16)(e - (float)h);
  red[d] = e * e;
  __syncthreads();
  for (int s = 128; s > 0; s >>= 1) {
    if (d < s) red[d] += red[d + s];
    __syncthreads();
  }
  if (d == 0) enorm[c] = red[0];
}

// ---------------------------------------------------------------------------
// Prep: z NCHW fp32 -> zf [N_VEC][EMB_DIM] bf16 hi/lo. One block per row n.
__global__ __launch_bounds__(256) void vq_prep_z(
    const float* __restrict__ z,
    __bf16* __restrict__ zhi, __bf16* __restrict__ zlo) {
  const int n = blockIdx.x;          // n = b*1024 + h*32 + w
  const int d = threadIdx.x;         // channel
  const int b = n >> 10, hw = n & 1023;
  const float v = z[((b * EMB_DIM + d) << 10) + hw];
  const __bf16 h = (__bf16)v;
  zhi[n * EMB_DIM + d] = h;
  zlo[n * EMB_DIM + d] = (__bf16)(v - (float)h);
}

// ---------------------------------------------------------------------------
// Main fused kernel: WMMA distance GEMM + argmin + z_q gather + loss partial.
// 256 threads = 8 waves; block owns rows n0..n0+15; wave owns 2048 codes.
__global__ __launch_bounds__(256) void vq_main(
    const float* __restrict__ z, const float* __restrict__ emb,
    const __bf16* __restrict__ ehi, const __bf16* __restrict__ elo,
    const __bf16* __restrict__ zhi, const __bf16* __restrict__ zlo,
    const float* __restrict__ enorm,
    float* __restrict__ zq_out, float* __restrict__ idx_out,
    float* __restrict__ loss_part) {
  __shared__ float ls_val[8 * 16 * 16];
  __shared__ int   ls_idx[8 * 16 * 16];
  __shared__ int   best_idx_s[16];
  __shared__ float red[256];

  const int tid   = threadIdx.x;
  const int wave  = tid >> 5;
  const int lane  = tid & 31;
  const int lhalf = lane & 15;
  const int hi16  = (lane >= 16) ? 1 : 0;
  const int n0    = blockIdx.x * 16;

  // ---- A tiles resident in registers: 16x32 A in contiguous k order (same as B) -----------
  // lanes 0-15: M=lane, K = ks..ks+15 contiguous
  // lanes 16-31: M=lane-16, K = ks+16..ks+31 contiguous
  v16bf a_hi[8], a_lo[8];
  {
    const __bf16* zh = zhi + (size_t)(n0 + lhalf) * EMB_DIM;
    const __bf16* zl = zlo + (size_t)(n0 + lhalf) * EMB_DIM;
#pragma unroll
    for (int s = 0; s < 8; ++s) {
      const int kb = s * 32 + (hi16 ? 16 : 0);
      v8bf h0 = *(const v8bf*)(zh + kb);
      v8bf h1 = *(const v8bf*)(zh + kb + 8);
      v8bf l0 = *(const v8bf*)(zl + kb);
      v8bf l1 = *(const v8bf*)(zl + kb + 8);
#pragma unroll
      for (int j = 0; j < 8; ++j) {
        a_hi[s][j] = h0[j]; a_hi[s][j + 8] = h1[j];
        a_lo[s][j] = l0[j]; a_lo[s][j + 8] = l1[j];
      }
    }
  }

  float minv[8];
  int   mini[8];
#pragma unroll
  for (int r = 0; r < 8; ++r) { minv[r] = 3.4e38f; mini[r] = 0; }

  const int cstart = wave * CODES_PER_WAVE;
  for (int t = 0; t < CODES_PER_WAVE / 16; ++t) {
    const int cbase = cstart + t * 16;
    const int c     = cbase + lhalf;
    // B 32x16 in contiguous k order (same as A): lanes 0-15: N=lane, K=ks..ks+15 contiguous;
    // lanes 16-31: N=lane-16, K=ks+16..ks+31 contiguous.
    const __bf16* ebh = ehi + (size_t)c * EMB_DIM + (hi16 ? 16 : 0);
    const __bf16* ebl = elo + (size_t)c * EMB_DIM + (hi16 ? 16 : 0);
    v8f acc = {};
#pragma unroll
    for (int s = 0; s < 8; ++s) {
      v16bf bh = *(const v16bf*)(ebh + s * 32);
      v16bf bl = *(const v16bf*)(ebl + s * 32);
      acc = __builtin_amdgcn_wmma_f32_16x16x32_bf16(false, a_hi[s], false, bh,
                                                    (short)0, acc, false, false);
      acc = __builtin_amdgcn_wmma_f32_16x16x32_bf16(false, a_hi[s], false, bl,
                                                    (short)0, acc, false, false);
      acc = __builtin_amdgcn_wmma_f32_16x16x32_bf16(false, a_lo[s], false, bh,
                                                    (short)0, acc, false, false);
    }
    const float en = enorm[c];
#pragma unroll
    for (int r = 0; r < 8; ++r) {
      const float dist = en - 2.0f * acc[r];
      if (dist < minv[r]) { minv[r] = dist; mini[r] = c; }  // '<' keeps first idx
    }
  }

  // ---- stash per-lane (val, idx); C layout: VGPR r -> row r (+8 for hi16) --
#pragma unroll
  for (int r = 0; r < 8; ++r) {
    const int row = r + (hi16 ? 8 : 0);
    const int off = wave * 256 + row * 16 + lhalf;
    ls_val[off] = minv[r];
    ls_idx[off] = mini[r];
  }
  __syncthreads();

  // ---- final argmin per row (16 rows x 128 candidates), first-index ties --
  if (tid < 16) {
    float bv = 3.4e38f; int bi = 0x7fffffff;
    for (int w2 = 0; w2 < 8; ++w2)
      for (int l2 = 0; l2 < 16; ++l2) {
        const int off = w2 * 256 + tid * 16 + l2;
        const float v = ls_val[off];
        const int   i = ls_idx[off];
        if (v < bv || (v == bv && i < bi)) { bv = v; bi = i; }
      }
    best_idx_s[tid] = bi;
    idx_out[n0 + tid] = (float)bi;
  }
  __syncthreads();

  // ---- z_q gather (fp32 codebook, exact), NCHW store, loss partial --------
  float part = 0.0f;
  for (int r = 0; r < 16; ++r) {
    const int n = n0 + r;
    const int b = n >> 10, hw = n & 1023;
    const int gi = best_idx_s[r];
    const float q  = emb[(size_t)gi * EMB_DIM + tid];
    const int zoff = ((b * EMB_DIM + tid) << 10) + hw;
    const float zv = z[zoff];
    zq_out[zoff] = q;                 // straight-through: value == z_q
    const float df = q - zv;
    part += df * df;
  }
  red[tid] = part;
  __syncthreads();
  for (int s = 128; s > 0; s >>= 1) {
    if (tid < s) red[tid] += red[tid + s];
    __syncthreads();
  }
  if (tid == 0) loss_part[blockIdx.x] = red[0];
}

// ---------------------------------------------------------------------------
// Deterministic fixed-order loss reduction: loss = 1.25 * mean((zq - z)^2)
__global__ void vq_loss_final(const float* __restrict__ loss_part,
                              float* __restrict__ loss_out) {
  if (threadIdx.x == 0 && blockIdx.x == 0) {
    float s = 0.0f;
    for (int i = 0; i < NBLK; ++i) s += loss_part[i];
    *loss_out = s * (1.25f / (float)ZQ_ELEMS);
  }
}

// ---------------------------------------------------------------------------
extern "C" void kernel_launch(void* const* d_in, const int* in_sizes, int n_in,
                              void* d_out, int out_size, void* d_ws, size_t ws_size,
                              hipStream_t stream) {
  const float* z   = (const float*)d_in[0];   // [8,256,32,32]
  const float* emb = (const float*)d_in[1];   // [16384,256]

  char* ws = (char*)d_ws;
  __bf16* ehi  = (__bf16*)(ws);                              //  8 MB
  __bf16* elo  = (__bf16*)(ws + ((size_t)8  << 20));         //  8 MB
  __bf16* zhi  = (__bf16*)(ws + ((size_t)16 << 20));         //  4 MB
  __bf16* zlo  = (__bf16*)(ws + ((size_t)20 << 20));         //  4 MB
  float* enorm = (float*) (ws + ((size_t)24 << 20));         // 64 KB
  float* lpart = (float*) (ws + ((size_t)24 << 20) + 65536); //  2 KB

  float* zq   = (float*)d_out;         // [2097152] z_q, NCHW
  float* loss = zq + ZQ_ELEMS;         // [1]
  float* idxo = loss + 1;              // [8192] indices as float values

  vq_prep_emb<<<N_CODES, 256, 0, stream>>>(emb, ehi, elo, enorm);
  vq_prep_z  <<<N_VEC,   256, 0, stream>>>(z, zhi, zlo);
  vq_main    <<<NBLK,    256, 0, stream>>>(z, emb, ehi, elo, zhi, zlo, enorm,
                                           zq, idxo, lpart);
  vq_loss_final<<<1, 64, 0, stream>>>(lpart, loss);
}
